// AngularPenaltySMLoss_72507637891693
// MI455X (gfx1250) — matrix operand order repaired and checked by reading, not compiled, not run on hardware
//
#include <hip/hip_runtime.h>
#include <hip/hip_bf16.h>

// CosFace loss fused pipeline for MI455X (gfx1250, wave32, WMMA).
// wf (4096 x 100000) is never materialized; the WMMA epilogue reduces
// exp(S*wf) per row. W stays L2-resident (f16 slab-converted on the fly).

typedef __attribute__((ext_vector_type(16))) _Float16 v16h;
typedef __attribute__((ext_vector_type(8)))  _Float16 v8h;
typedef __attribute__((ext_vector_type(8)))  float    v8f;

#define D_DIM      512
#define S_SCALE    30.0f
#define MARGIN     0.35f
#define LDS_STRIDE 40      // 32 f16 of K + 8 f16 pad per column (80 B, 16B-aligned slots)
#define BCOLS      128     // B slab width (columns per LDS stage)
#define NTILE      8       // WMMA column tiles per wave per K-step

// ds_swizzle_b32 XOR-butterfly add (group-of-32 mode: offset = xor<<10 | and 0x1f)
template <int MASK>
__device__ __forceinline__ float swz_add(float s) {
  return s + __int_as_float(__builtin_amdgcn_ds_swizzle(__float_as_int(s), MASK));
}
__device__ __forceinline__ float half16_sum(float s) {
  // reduction within each 16-lane half of a wave32
  s = swz_add<0x041F>(s);   // SWAPX1
  s = swz_add<0x081F>(s);   // SWAPX2
  s = swz_add<0x101F>(s);   // SWAPX4
  s = swz_add<0x201F>(s);   // SWAPX8
  return s;
}
__device__ __forceinline__ float wave32_sum(float s) {
  s = half16_sum(s);
  s = swz_add<0x401F>(s);   // SWAPX16
  return s;
}

__device__ __forceinline__ v8h cvt8(float4 w0, float4 w1) {
  v8h h;
  h[0] = (_Float16)w0.x; h[1] = (_Float16)w0.y;
  h[2] = (_Float16)w0.z; h[3] = (_Float16)w0.w;
  h[4] = (_Float16)w1.x; h[5] = (_Float16)w1.y;
  h[6] = (_Float16)w1.z; h[7] = (_Float16)w1.w;
  return h;
}

// ---------------------------------------------------------------- rownorm ---
__global__ __launch_bounds__(256) void rownorm_kernel(
    const float* __restrict__ x, float* __restrict__ invn, int N) {
  const int wave = threadIdx.x >> 5;
  const int lane = threadIdx.x & 31;
  const int row  = blockIdx.x * 8 + wave;
  if (row >= N) return;                       // wave-uniform
  const float4* xr = (const float4*)(x + (size_t)row * D_DIM);
  float s = 0.0f;
#pragma unroll
  for (int i = 0; i < 4; ++i) {               // 512 floats / 32 lanes = 4 float4 each
    float4 v = xr[lane + 32 * i];
    s += v.x * v.x + v.y * v.y + v.z * v.z + v.w * v.w;
  }
  s = wave32_sum(s);
  if (lane == 0) invn[row] = 1.0f / fmaxf(sqrtf(s), 1e-12f);
}

// ----------------------------------------------------------- target logit ---
__global__ __launch_bounds__(256) void target_kernel(
    const float* __restrict__ x, const float* __restrict__ W,
    const int* __restrict__ labels, const float* __restrict__ invn,
    float* __restrict__ tgt, int N) {
  const int wave = threadIdx.x >> 5;
  const int lane = threadIdx.x & 31;
  const int row  = blockIdx.x * 8 + wave;
  if (row >= N) return;                       // wave-uniform
  const int lbl = labels[row];
  const float4* xr = (const float4*)(x + (size_t)row * D_DIM);
  const float4* wr = (const float4*)(W + (size_t)lbl * D_DIM);
  float s = 0.0f;
#pragma unroll
  for (int i = 0; i < 4; ++i) {
    float4 a = xr[lane + 32 * i];
    float4 b = wr[lane + 32 * i];
    s += a.x * b.x + a.y * b.y + a.z * b.z + a.w * b.w;
  }
  s = wave32_sum(s);
  if (lane == 0) tgt[row] = s * invn[row];
}

// -------------------------------------------------- fused WMMA GEMM + exp ---
// Block: 256 threads = 8 waves. Block tile: 128 rows x (2 chunks x 128 cols).
// Each wave owns 16 rows and 8 WMMA accumulators (16x128 of wf per chunk):
// one A fragment feeds 8 back-to-back WMMAs, halving A-side L2 re-reads.
// Double-buffered LDS: one barrier per K-step; next W slab loads overlap WMMA.
__global__ __launch_bounds__(256) void cosface_gemm_kernel(
    const float* __restrict__ x, const float* __restrict__ W,
    const float* __restrict__ invn, float* __restrict__ rowsum,
    int N, int C) {
  __shared__ __align__(16) _Float16 lb[2][BCOLS * LDS_STRIDE]; // 2 x (128 cols x 32 K)

  const int tid  = threadIdx.x;
  const int wave = tid >> 5;
  const int lane = tid & 31;

  const int row0 = blockIdx.y * 128 + wave * 16;
  int rowA       = row0 + (lane & 15);
  if (rowA > N - 1) rowA = N - 1;             // safety clamp (N=4096 divides exactly)
  const float4* xr = (const float4*)(x + (size_t)rowA * D_DIM);

  // Normalization folded into the epilogue: wf = inv_row * (x . W_col).
  // Per-lane epilogue row for accumulator VGPR v is row0 + v + 8*(lane>=16).
  float sein[8];
#pragma unroll
  for (int v = 0; v < 8; ++v) {
    int r = row0 + v + ((lane >> 4) << 3);
    sein[v] = S_SCALE * invn[r < N ? r : N - 1];
  }

  // cooperative B-stage mapping: 2 threads per column, 16 K-values each
  const int tcol = tid >> 1;                  // 0..127 local column
  const int khex = (tid & 1) * 16;            // K-half within 32-wide slab
  const int lds_wr = tcol * LDS_STRIDE + khex;

  float sv[8] = {0, 0, 0, 0, 0, 0, 0, 0};    // per-lane partial sums of exp(S*wf)

  for (int chunk = 0; chunk < 2; ++chunk) {
    const int c0 = blockIdx.x * 256 + chunk * BCOLS;
    if (c0 >= C) break;                       // block-uniform
    const int cglob = c0 + tcol;
    const int cc = (cglob < C) ? cglob : (C - 1);
    const float4* wr = (const float4*)(W + (size_t)cc * D_DIM);

    v8f acc[NTILE] = {};                      // 16x128 fp32 accumulators

    // prologue: stage slab kb=0 into buffer 0 (prev chunk's last reads were
    // from buffer 1 and are fenced by the final barrier of its K-loop)
    {
      const int kq0 = khex >> 2;
      *(v8h*)&lb[0][lds_wr]     = cvt8(wr[kq0],     wr[kq0 + 1]);
      *(v8h*)&lb[0][lds_wr + 8] = cvt8(wr[kq0 + 2], wr[kq0 + 3]);
    }

    int p = 0;
    for (int kb = 0; kb < D_DIM; kb += 32) {
      __syncthreads();                        // slab p staged; slab p^1 readers done

      // ---- issue next slab's global loads early (overlap with WMMA) ----
      const bool have_next = (kb + 32) < D_DIM;
      float4 nw0 = {}, nw1 = {}, nw2 = {}, nw3 = {};
      if (have_next) {
        const int kq = (kb + 32 + khex) >> 2;
        nw0 = wr[kq];     nw1 = wr[kq + 1];
        nw2 = wr[kq + 2]; nw3 = wr[kq + 3];
        __builtin_prefetch(&wr[kq + 8], 0, 1);   // pull slab after next toward cache
      }

      // ---- A fragment: 16x32 f16, contiguous layout (lanes 0-15: K kb..+15;
      //      lanes 16-31: K kb+16..+31), raw x (normalize in epilogue) ----
      const int aq = (kb >> 2) + ((lane >> 4) << 2);
      float4 a0 = xr[aq], a1 = xr[aq + 1], a2 = xr[aq + 2], a3 = xr[aq + 3];
      v16h A;
      A[0]  = (_Float16)a0.x; A[1]  = (_Float16)a0.y;
      A[2]  = (_Float16)a0.z; A[3]  = (_Float16)a0.w;
      A[4]  = (_Float16)a1.x; A[5]  = (_Float16)a1.y;
      A[6]  = (_Float16)a1.z; A[7]  = (_Float16)a1.w;
      A[8]  = (_Float16)a2.x; A[9]  = (_Float16)a2.y;
      A[10] = (_Float16)a2.z; A[11] = (_Float16)a2.w;
      A[12] = (_Float16)a3.x; A[13] = (_Float16)a3.y;
      A[14] = (_Float16)a3.z; A[15] = (_Float16)a3.w;

      // ---- 8 column tiles: B fragment = 16 contiguous K f16 per lane ----
#pragma unroll
      for (int t = 0; t < NTILE; ++t) {
        const int off = (t * 16 + (lane & 15)) * LDS_STRIDE + ((lane >> 4) << 4);
        v8h b0 = *(const v8h*)&lb[p][off];
        v8h b1 = *(const v8h*)&lb[p][off + 8];
        v16h B = __builtin_shufflevector(b0, b1, 0, 1, 2, 3, 4, 5, 6, 7,
                                                 8, 9, 10, 11, 12, 13, 14, 15);
        acc[t] = __builtin_amdgcn_wmma_f32_16x16x32_f16(
            false, A, false, B, (short)0, acc[t], false, false);
      }

      // ---- stage next slab into the other buffer ----
      if (have_next) {
        *(v8h*)&lb[p ^ 1][lds_wr]     = cvt8(nw0, nw1);
        *(v8h*)&lb[p ^ 1][lds_wr + 8] = cvt8(nw2, nw3);
      }
      p ^= 1;
    }

    // ---- epilogue: masked exp(S*inv*dot), accumulate per (lane, VGPR-row) ----
    const int ncol = lane & 15;
#pragma unroll
    for (int t = 0; t < NTILE; ++t) {
      const bool valid = (c0 + t * 16 + ncol) < C;
#pragma unroll
      for (int v = 0; v < 8; ++v) {
        sv[v] += valid ? __expf(sein[v] * acc[t][v]) : 0.0f;
      }
    }
  }

  // ---- per-row reduction: lanes 0-15 hold row v, lanes 16-31 hold row v+8 ----
#pragma unroll
  for (int v = 0; v < 8; ++v) {
    float s = half16_sum(sv[v]);
    if (lane == 0  && (row0 + v)     < N) atomicAdd(&rowsum[row0 + v], s);
    if (lane == 16 && (row0 + v + 8) < N) atomicAdd(&rowsum[row0 + v + 8], s);
  }
}

// --------------------------------------------------------------- finalize ---
__global__ __launch_bounds__(256) void finalize_kernel(
    const float* __restrict__ tgt, const float* __restrict__ rowsum,
    float* __restrict__ out, int N) {
  __shared__ float red[256];
  float acc = 0.0f;
  for (int i = threadIdx.x; i < N; i += 256) {
    const float t   = tgt[i];
    const float num = S_SCALE * (t - MARGIN);
    const float den = __expf(num) + rowsum[i] - __expf(S_SCALE * t);
    acc += num - __logf(den);
  }
  red[threadIdx.x] = acc;
  __syncthreads();
  for (int s = 128; s > 0; s >>= 1) {
    if (threadIdx.x < s) red[threadIdx.x] += red[threadIdx.x + s];
    __syncthreads();
  }
  if (threadIdx.x == 0) out[0] = -red[0] / (float)N;
}

// ------------------------------------------------------------------ launch ---
extern "C" void kernel_launch(void* const* d_in, const int* in_sizes, int n_in,
                              void* d_out, int out_size, void* d_ws, size_t ws_size,
                              hipStream_t stream) {
  (void)n_in; (void)out_size; (void)ws_size;
  const float* x      = (const float*)d_in[0];   // [N, 512] f32
  const float* W      = (const float*)d_in[1];   // [C, 512] f32
  const int*   labels = (const int*)d_in[2];     // [N] i32
  float* out = (float*)d_out;                    // scalar f32

  const int N = in_sizes[2];
  const int C = in_sizes[1] / D_DIM;

  float* ws     = (float*)d_ws;                  // < 64 KB scratch
  float* invn   = ws;                            // [N]
  float* tgt    = ws + N;                        // [N]
  float* rowsum = ws + 2 * N;                    // [N]

  hipMemsetAsync(rowsum, 0, (size_t)N * sizeof(float), stream);

  const int rowBlocks = (N + 7) / 8;
  rownorm_kernel<<<rowBlocks, 256, 0, stream>>>(x, invn, N);
  target_kernel<<<rowBlocks, 256, 0, stream>>>(x, W, labels, invn, tgt, N);

  dim3 grid((C + 255) / 256, (N + 127) / 128);
  cosface_gemm_kernel<<<grid, 256, 0, stream>>>(x, W, invn, rowsum, N, C);

  finalize_kernel<<<1, 256, 0, stream>>>(tgt, rowsum, out, N);
}
